// PPI_Model_25786983646091
// MI455X (gfx1250) — matrix operand order repaired and checked by reading, not compiled, not run on hardware
//
#include <hip/hip_runtime.h>
#include <hip/hip_fp16.h>
#include <math.h>

typedef __attribute__((ext_vector_type(16))) _Float16 v16h;
typedef __attribute__((ext_vector_type(8)))  float    v8f;

#define N_RES   384
#define DIMF    115
#define H1W     462
#define H1P     480      // 462 padded to 15*32
#define KSTEPS  15
#define MD      16
#define NA      3072
#define NE      24576
#define NPOS    38       // int(384*0.1)
#define DEPTH   4

__device__ __forceinline__ float siluf(float x) { return x / (1.f + __expf(-x)); }
__device__ __forceinline__ float sigf(float x)  { return 1.f / (1.f + __expf(-x)); }

// ---------------- generic helpers ----------------
__global__ void k_zero(float* p, int n) {
    int i = blockIdx.x * blockDim.x + threadIdx.x;
    if (i < n) p[i] = 0.f;
}
__global__ void k_copy(float* dst, const float* src, int n) {
    int i = blockIdx.x * blockDim.x + threadIdx.x;
    if (i < n) dst[i] = src[i];
}

// ---------------- batchnorm(bert_var) + concat, coords init ----------------
__global__ void k_bn(const float* nf, const float* bv, const float* pd,
                     float* feats, float* coors) {
    int i = blockIdx.x, t = threadIdx.x;            // block = 64
    __shared__ float red[64];
    float v = bv[i * 64 + t];
    red[t] = v; __syncthreads();
    for (int s = 32; s > 0; s >>= 1) { if (t < s) red[t] += red[t + s]; __syncthreads(); }
    float mu = red[0] / 64.f; __syncthreads();
    float d = v - mu; red[t] = d * d; __syncthreads();
    for (int s = 32; s > 0; s >>= 1) { if (t < s) red[t] += red[t + s]; __syncthreads(); }
    float var = red[0] / 64.f;
    feats[i * DIMF + 51 + t] = (v - mu) * rsqrtf(var + 1e-5f);
    if (t < 51) feats[i * DIMF + t] = nf[i * 51 + t];
    if (t < 3)  coors[i * 3 + t] = pd[i * 3 + t];
}

// ---------------- Fi = feats@W1[0:115], Fj = feats@W1[115:230] ----------------
__global__ void k_fifj(const float* feats, const float* w1, float* Fi, float* Fj) {
    int idx = blockIdx.x * blockDim.x + threadIdx.x;
    const int total = N_RES * H1W;
    if (idx >= 2 * total) return;
    int which = idx / total, r = idx % total;
    int i = r / H1W, o = r % H1W;
    const float* W = w1 + (which ? DIMF * H1W : 0);
    const float* f = feats + i * DIMF;
    float acc = 0.f;
    for (int k = 0; k < DIMF; k++) acc += f[k] * W[k * H1W + o];
    (which ? Fj : Fi)[i * H1W + o] = acc;
}

// ------ swizzle e2_w (462x16) and c1_w (16x64) into WMMA B-fragment order ------
__global__ void k_convw(const float* e2w, const float* c1w,
                        _Float16* e2f, _Float16* c1f) {
    int idx = blockIdx.x * blockDim.x + threadIdx.x;
    if (idx < KSTEPS * 32 * 16) {
        int t = idx & 15, lane = (idx >> 4) & 31, ks = idx >> 9;
        int k = ks * 32 + ((t >> 3) * 16) + ((lane >> 4) * 8) + (t & 7), nn = lane & 15;
        e2f[idx] = (k < H1W) ? (_Float16)e2w[k * 16 + nn] : (_Float16)0.f;
    } else if (idx < KSTEPS * 32 * 16 + 4 * 32 * 16) {
        int j = idx - KSTEPS * 32 * 16;
        int t = j & 15, lane = (j >> 4) & 31, nt = j >> 9;
        int k = (t >> 3) * 16 + (lane >> 4) * 8 + (t & 7), nn = nt * 16 + (lane & 15);
        c1f[j] = (k < MD) ? (_Float16)c1w[k * 64 + nn] : (_Float16)0.f;
    }
}

// ---------------- the heavy edge kernel: one wave per (i, 16 j) tile ----------------
__global__ void __launch_bounds__(32)
k_edge(const float* __restrict__ Fi, const float* __restrict__ Fj,
       const float* __restrict__ coors,
       const float* __restrict__ wd, const float* __restrict__ e1b,
       const _Float16* __restrict__ e2frag, const float* __restrict__ e2b,
       const _Float16* __restrict__ c1frag, const float* __restrict__ c1b,
       const float* __restrict__ c2w, const float* __restrict__ c2b,
       const float* __restrict__ cns,
       float* __restrict__ m_i, float* __restrict__ coors_out) {
    __shared__ _Float16 h1[16][H1P];
    __shared__ float fi_s[H1P];
    __shared__ float d2s[16];
    __shared__ float mbuf[16][16];
    __shared__ _Float16 mh[16][16];
    __shared__ float h2[16][64];
    __shared__ float cwred[16][3];

    int lane = threadIdx.x;
    int i = blockIdx.y, j0 = blockIdx.x * 16;

    if (lane < 16) __builtin_prefetch(Fj + (size_t)(j0 + lane) * H1W, 0, 1);

    float cix = coors[i * 3 + 0], ciy = coors[i * 3 + 1], ciz = coors[i * 3 + 2];
    for (int k = lane; k < H1P; k += 32) fi_s[k] = (k < H1W) ? Fi[i * H1W + k] : 0.f;
    if (lane < 16) {
        int j = j0 + lane;
        float dx = cix - coors[j * 3 + 0], dy = ciy - coors[j * 3 + 1], dz = ciz - coors[j * 3 + 2];
        d2s[lane] = dx * dx + dy * dy + dz * dz;
    }
    __syncthreads();

    // h1[j][k] = silu(Fi[i,k] + Fj[j,k] + d2*wd[k] + b[k])  (f16, zero-padded K)
    for (int j = 0; j < 16; j++) {
        float d2 = d2s[j];
        const float* fj = Fj + (size_t)(j0 + j) * H1W;
        for (int k = lane; k < H1P; k += 32) {
            float v = 0.f;
            if (k < H1W) v = siluf(fi_s[k] + fj[k] + d2 * wd[k] + e1b[k]);
            h1[j][k] = (_Float16)v;
        }
    }
    __syncthreads();

    // GEMM1: m_pre[16j x 16md] = h1[16 x 480] @ e2frag  via 15 WMMA k-steps
    v8f acc = {};
    int row = lane & 15, koff = (lane >> 4) * 8;
    for (int ks = 0; ks < KSTEPS; ks++) {
        v16h a, b;
        int kb = ks * 32;
#pragma unroll
        for (int t = 0; t < 8; t++) {
            a[t]     = h1[row][kb + koff + t];
            a[t + 8] = h1[row][kb + 16 + koff + t];
        }
        const _Float16* bp = e2frag + (size_t)(ks * 32 + lane) * 16;
#pragma unroll
        for (int t = 0; t < 16; t++) b[t] = bp[t];
        acc = __builtin_amdgcn_wmma_f32_16x16x32_f16(false, a, false, b,
                                                     (short)0, acc, false, false);
    }
    {   // m = silu(m_pre + e2_b);  D layout: lane&15 = N(col), vgpr v = M row (+8 for hi lanes)
        int col = lane & 15, rbase = (lane >> 4) * 8;
#pragma unroll
        for (int v = 0; v < 8; v++) {
            float x = siluf(acc[v] + e2b[col]);
            mbuf[rbase + v][col] = x;
            mh[rbase + v][col] = (_Float16)x;
        }
    }
    __syncthreads();

    // m_i[i] += sum_j m[j]
    if (lane < 16) {
        float s = 0.f;
        for (int j = 0; j < 16; j++) s += mbuf[j][lane];
        atomicAdd(&m_i[i * MD + lane], s);
    }

    // GEMM2: h2[16j x 64] = silu(m @ c1_w + c1_b), K=16 zero-padded to 32
    v16h a2;
#pragma unroll
    for (int t = 0; t < 8; t++) { a2[t] = mh[row][koff + t]; a2[t + 8] = (_Float16)0.f; }
    for (int nt = 0; nt < 4; nt++) {
        v16h b2;
        const _Float16* bp = c1frag + (size_t)(nt * 32 + lane) * 16;
#pragma unroll
        for (int t = 0; t < 16; t++) b2[t] = bp[t];
        v8f cacc = {};
        cacc = __builtin_amdgcn_wmma_f32_16x16x32_f16(false, a2, false, b2,
                                                      (short)0, cacc, false, false);
        int col = nt * 16 + (lane & 15), rbase = (lane >> 4) * 8;
#pragma unroll
        for (int v = 0; v < 8; v++) h2[rbase + v][col] = siluf(cacc[v] + c1b[col]);
    }
    __syncthreads();

    // cw = clamp(h2 @ c2_w + c2_b, ±2); coors update with CoorsNorm
    if (lane < 16) {
        float s = c2b[0];
        for (int t = 0; t < 64; t++) s += h2[lane][t] * c2w[t];
        s = fminf(fmaxf(s, -2.f), 2.f);
        int j = j0 + lane;
        float dx = cix - coors[j * 3 + 0], dy = ciy - coors[j * 3 + 1], dz = ciz - coors[j * 3 + 2];
        float nrm = sqrtf(dx * dx + dy * dy + dz * dz);
        float inv = cns[0] / fmaxf(nrm, 1e-8f);
        cwred[lane][0] = s * dx * inv;
        cwred[lane][1] = s * dy * inv;
        cwred[lane][2] = s * dz * inv;
    }
    __syncthreads();
    if (lane < 3) {
        float s = 0.f;
        for (int j = 0; j < 16; j++) s += cwred[j][lane];
        atomicAdd(&coors_out[i * 3 + lane], s);
    }
}

// ---------------- node MLP: feats += silu([feats,m_i]@n1+b)@n2 + b2 ----------------
__global__ void k_node1(const float* feats, const float* m_i,
                        const float* w, const float* b, float* tmp) {
    int idx = blockIdx.x * blockDim.x + threadIdx.x;
    if (idx >= N_RES * 230) return;
    int i = idx / 230, o = idx % 230;
    float acc = b[o];
    const float* f = feats + i * DIMF;
    for (int k = 0; k < DIMF; k++) acc += f[k] * w[k * 230 + o];
    const float* m = m_i + i * MD;
    for (int k = 0; k < MD; k++) acc += m[k] * w[(DIMF + k) * 230 + o];
    tmp[idx] = siluf(acc);
}
__global__ void k_node2(const float* feats, const float* tmp,
                        const float* w, const float* b, float* out) {
    int idx = blockIdx.x * blockDim.x + threadIdx.x;
    if (idx >= N_RES * DIMF) return;
    int i = idx / DIMF, o = idx % DIMF;
    float acc = b[o];
    const float* t = tmp + i * 230;
    for (int k = 0; k < 230; k++) acc += t[k] * w[k * DIMF + o];
    out[idx] = feats[idx] + acc;
}

// ---------------- atom model ----------------
__global__ void k_atom_embed(const float* oh, const float* w, const float* b, float* x) {
    int idx = blockIdx.x * blockDim.x + threadIdx.x;
    if (idx >= NA * DIMF) return;
    int a = idx / DIMF, o = idx % DIMF;
    float acc = b[o];
    const float* r = oh + a * 37;
    for (int k = 0; k < 37; k++) acc += r[k] * w[k * DIMF + o];
    x[idx] = acc;
}
__global__ void k_deg(const int* dst, float* deg) {
    int e = blockIdx.x * blockDim.x + threadIdx.x;
    if (e < NE) atomicAdd(&deg[dst[e]], 1.f);
}
__global__ void k_scatter(const int* src, const int* dst, const float* x, float* agg) {
    int idx = blockIdx.x * blockDim.x + threadIdx.x;
    if (idx >= NE * DIMF) return;
    int e = idx / DIMF, f = idx % DIMF;
    atomicAdd(&agg[dst[e] * DIMF + f], x[src[e] * DIMF + f]);
}
__global__ void k_sage(const float* x, const float* agg, const float* deg,
                       const float* wl, const float* bl, const float* wr, float* xo) {
    int idx = blockIdx.x * blockDim.x + threadIdx.x;
    if (idx >= NA * DIMF) return;
    int a = idx / DIMF, o = idx % DIMF;
    float dinv = 1.f / fmaxf(deg[a], 1.f);
    float acc = bl[o];
    const float* ar = agg + a * DIMF;
    const float* xr = x + a * DIMF;
    for (int k = 0; k < DIMF; k++) acc += (ar[k] * dinv) * wl[k * DIMF + o] + xr[k] * wr[k * DIMF + o];
    xo[idx] = acc;
}
__global__ void k_res_scatter(const float* x, const int* map, float* fr) {
    int idx = blockIdx.x * blockDim.x + threadIdx.x;
    if (idx >= NA * DIMF) return;
    int a = idx / DIMF, f = idx % DIMF;
    atomicAdd(&fr[map[a] * DIMF + f], x[idx]);
}

// ---------------- contrastive projector ----------------
__global__ void k_proj1(const float* z, const float* w, const float* b, float* tmp) {
    int idx = blockIdx.x * blockDim.x + threadIdx.x;
    if (idx >= N_RES * DIMF) return;
    int i = idx / DIMF, o = idx % DIMF;
    float acc = b[o];
    const float* r = z + i * DIMF;
    for (int k = 0; k < DIMF; k++) acc += r[k] * w[k * DIMF + o];
    tmp[idx] = (acc > 0.f) ? acc : (__expf(acc) - 1.f);
}
__global__ void k_proj2(const float* tmp, const float* w, const float* b, float* out) {
    int idx = blockIdx.x * blockDim.x + threadIdx.x;
    if (idx >= N_RES * DIMF) return;
    int i = idx / DIMF, o = idx % DIMF;
    float acc = b[o];
    const float* r = tmp + i * DIMF;
    for (int k = 0; k < DIMF; k++) acc += r[k] * w[k * DIMF + o];
    out[idx] = acc;
}

// ---------------- meta-pos mask: 38 smallest nonzero per row + diag ----------------
__global__ void k_mask(const float* adj, float* mask) {
    int i = blockIdx.x;
    if (threadIdx.x != 0) return;
    float lastv = -1e30f; int lastidx = -1;
    for (int s = 0; s < NPOS; s++) {
        float best = 3.4e38f; int bi = -1;
        for (int j = 0; j < N_RES; j++) {
            float v = adj[i * N_RES + j];
            if (v == 0.f) continue;
            if (v < lastv || (v == lastv && j <= lastidx)) continue;
            if (v < best) { best = v; bi = j; }
        }
        if (bi < 0) break;
        mask[i * N_RES + bi] = 1.f;
        lastv = best; lastidx = bi;
    }
    mask[i * N_RES + i] = 1.f;
}

// ---------------- similarity + InfoNCE losses ----------------
__global__ void k_norms(const float* mp, const float* sp, float* norms) {
    int idx = blockIdx.x * blockDim.x + threadIdx.x;
    if (idx >= 2 * N_RES) return;
    const float* r = (idx < N_RES ? mp + idx * DIMF : sp + (idx - N_RES) * DIMF);
    float s = 0.f;
    for (int k = 0; k < DIMF; k++) s += r[k] * r[k];
    norms[idx] = sqrtf(s);
}
__global__ void k_sim(const float* mp, const float* sp, const float* norms, float* sim) {
    int idx = blockIdx.x * blockDim.x + threadIdx.x;
    if (idx >= N_RES * N_RES) return;
    int i = idx / N_RES, j = idx % N_RES;
    const float* a = mp + i * DIMF; const float* b = sp + j * DIMF;
    float d = 0.f;
    for (int k = 0; k < DIMF; k++) d += a[k] * b[k];
    sim[idx] = __expf(d / (norms[i] * norms[N_RES + j]) / 0.8f);
}
__global__ void k_loss(const float* sim, const float* mask, float* out) {
    int i = blockIdx.x, t = threadIdx.x;               // block = 128
    __shared__ float s0[128], s1[128], s2[128], s3[128];
    float rs = 0, rm = 0, cs = 0, cm = 0;
    for (int j = t; j < N_RES; j += 128) {
        float sv = sim[i * N_RES + j], mv = mask[i * N_RES + j];
        rs += sv; rm += sv * mv;
        float cv = sim[j * N_RES + i];
        cs += cv; cm += cv * mv;                       // mask[i][j] for column loss of row i
    }
    s0[t] = rs; s1[t] = rm; s2[t] = cs; s3[t] = cm; __syncthreads();
    for (int s = 64; s > 0; s >>= 1) {
        if (t < s) { s0[t] += s0[t+s]; s1[t] += s1[t+s]; s2[t] += s2[t+s]; s3[t] += s3[t+s]; }
        __syncthreads();
    }
    if (t == 0) {
        float tm = -__logf(s1[0] / (s0[0] + 1e-8f));
        float ts = -__logf(s3[0] / (s2[0] + 1e-8f));
        atomicAdd(out, 0.5f * (tm + ts) / (float)N_RES);
    }
}

// ---------------- head: attention(seq=1 -> ctx = k-proj) + MLP ----------------
__global__ void k_head(const float* mp, const float* sp,
                       const float* kw, const float* gw, const float* gb,
                       const float* m1w, const float* m1b,
                       const float* m2w, const float* m2b,
                       const float* m3w, const float* m3b, float* out) {
    int i = blockIdx.x, t = threadIdx.x;               // block = 256
    __shared__ float emb[230], att[230], h1[128], h2[16];
    if (t < DIMF) { emb[t] = mp[i * DIMF + t]; emb[DIMF + t] = sp[i * DIMF + t]; }
    __syncthreads();
    for (int o = t; o < 230; o += 256) {
        float kv = 0.f, g = gb[o];
        for (int k = 0; k < 230; k++) { kv += emb[k] * kw[k * 230 + o]; g += emb[k] * gw[k * 230 + o]; }
        att[o] = sigf(g) * kv;
    }
    __syncthreads();
    for (int o = t; o < 128; o += 256) {
        float a = m1b[o];
        for (int k = 0; k < 230; k++) a += att[k] * m1w[k * 128 + o];
        h1[o] = fmaxf(a, 0.f);
    }
    __syncthreads();
    if (t < 16) {
        float a = m2b[t];
        for (int k = 0; k < 128; k++) a += h1[k] * m2w[k * 16 + t];
        h2[t] = fmaxf(a, 0.f);
    }
    __syncthreads();
    if (t == 0) {
        float a = m3b[0];
        for (int k = 0; k < 16; k++) a += h2[k] * m3w[k];
        out[i] = sigf(a);
    }
}

// ==================================================================
extern "C" void kernel_launch(void* const* d_in, const int* in_sizes, int n_in,
                              void* d_out, int out_size, void* d_ws, size_t ws_size,
                              hipStream_t stream) {
    // inputs in setup_inputs() dict order; params expanded in insertion order
    const float* nf     = (const float*)d_in[1];
    const float* bv     = (const float*)d_in[2];
    const float* pd     = (const float*)d_in[3];
    const float* adj    = (const float*)d_in[4];
    const float* onehot = (const float*)d_in[5];
    const int*   eidx   = (const int*)d_in[6];
    const int*   r2a    = (const int*)d_in[7];
    const float* e1w = (const float*)d_in[9],  *e1b = (const float*)d_in[10];
    const float* e2w = (const float*)d_in[11], *e2b = (const float*)d_in[12];
    const float* c1w = (const float*)d_in[13], *c1bp = (const float*)d_in[14];
    const float* c2w = (const float*)d_in[15], *c2bp = (const float*)d_in[16];
    const float* cns = (const float*)d_in[17];
    const float* n1w = (const float*)d_in[18], *n1b = (const float*)d_in[19];
    const float* n2w = (const float*)d_in[20], *n2b = (const float*)d_in[21];
    const float* lnw = (const float*)d_in[22], *lnb = (const float*)d_in[23];
    const float* swl = (const float*)d_in[24], *sbl = (const float*)d_in[25];
    const float* swr = (const float*)d_in[26];
    const float* p1w = (const float*)d_in[27], *p1b = (const float*)d_in[28];
    const float* p2w = (const float*)d_in[29], *p2b = (const float*)d_in[30];
    const float* kw  = (const float*)d_in[32];
    const float* gw  = (const float*)d_in[33], *gb  = (const float*)d_in[34];
    const float* m1w = (const float*)d_in[35], *m1b = (const float*)d_in[36];
    const float* m2w = (const float*)d_in[37], *m2b = (const float*)d_in[38];
    const float* m3w = (const float*)d_in[39], *m3b = (const float*)d_in[40];
    float* out = (float*)d_out;

    float* ws = (float*)d_ws;
    size_t off = 0;
    auto alloc = [&](size_t n) { float* p = ws + off; off += n; return p; };
    float* f0    = alloc(N_RES * DIMF);
    float* f1    = alloc(N_RES * DIMF);
    float* cr0   = alloc(N_RES * 3);
    float* cr1   = alloc(N_RES * 3);
    float* Fi    = alloc(N_RES * H1W);   // also reused as tmp230 for node MLP
    float* Fj    = alloc(N_RES * H1W);   // also reused as projector tmp
    float* m_i   = alloc(N_RES * MD);
    _Float16* e2frag = (_Float16*)alloc(KSTEPS * 32 * 16 / 2 + 8);
    _Float16* c1frag = (_Float16*)alloc(4 * 32 * 16 / 2 + 8);
    float* xa    = alloc(NA * DIMF);
    float* xb    = alloc(NA * DIMF);
    float* agg   = alloc(NA * DIMF);
    float* deg   = alloc(NA);
    float* fres  = alloc(N_RES * DIMF);
    float* mp    = alloc(N_RES * DIMF);
    float* sp    = alloc(N_RES * DIMF);
    float* norms = alloc(2 * N_RES);
    float* simb  = alloc(N_RES * N_RES);
    float* maskb = alloc(N_RES * N_RES);

    auto blks = [](int n, int b) { return (n + b - 1) / b; };

    // feats / coors init
    k_bn<<<N_RES, 64, 0, stream>>>(nf, bv, pd, f0, cr0);

    // ---- EGNN depth 4 ----
    float* fcur = f0; float* fnxt = f1;
    float* ccur = cr0; float* cnxt = cr1;
    for (int l = 0; l < DEPTH; l++) {
        const float* w1 = e1w + (size_t)l * 231 * H1W;
        k_fifj<<<blks(2 * N_RES * H1W, 256), 256, 0, stream>>>(fcur, w1, Fi, Fj);
        k_convw<<<blks((KSTEPS + 4) * 32 * 16, 256), 256, 0, stream>>>(
            e2w + (size_t)l * H1W * MD, c1w + (size_t)l * MD * 64, e2frag, c1frag);
        k_copy<<<blks(N_RES * 3, 256), 256, 0, stream>>>(cnxt, ccur, N_RES * 3);
        k_zero<<<blks(N_RES * MD, 256), 256, 0, stream>>>(m_i, N_RES * MD);
        dim3 eg(N_RES / 16, N_RES);
        k_edge<<<eg, 32, 0, stream>>>(Fi, Fj, ccur,
                                      w1 + 230 * H1W, e1b + (size_t)l * H1W,
                                      e2frag, e2b + l * MD,
                                      c1frag, c1bp + l * 64,
                                      c2w + l * 64, c2bp + l, cns + l,
                                      m_i, cnxt);
        k_node1<<<blks(N_RES * 230, 256), 256, 0, stream>>>(
            fcur, m_i, n1w + (size_t)l * 131 * 230, n1b + l * 230, Fi);
        k_node2<<<blks(N_RES * DIMF, 256), 256, 0, stream>>>(
            fcur, Fi, n2w + (size_t)l * 230 * DIMF, n2b + l * DIMF, fnxt);
        float* tf = fcur; fcur = fnxt; fnxt = tf;
        float* tc = ccur; ccur = cnxt; cnxt = tc;
    }
    // fcur = egnn_output

    // ---- atom model ----
    k_atom_embed<<<blks(NA * DIMF, 256), 256, 0, stream>>>(onehot, lnw, lnb, xa);
    k_zero<<<blks(NA, 256), 256, 0, stream>>>(deg, NA);
    k_deg<<<blks(NE, 256), 256, 0, stream>>>(eidx + NE, deg);
    float* xc = xa; float* xn = xb;
    for (int l = 0; l < DEPTH; l++) {
        k_zero<<<blks(NA * DIMF, 256), 256, 0, stream>>>(agg, NA * DIMF);
        k_scatter<<<blks(NE * DIMF, 256), 256, 0, stream>>>(eidx, eidx + NE, xc, agg);
        k_sage<<<blks(NA * DIMF, 256), 256, 0, stream>>>(
            xc, agg, deg, swl + (size_t)l * DIMF * DIMF, sbl + l * DIMF,
            swr + (size_t)l * DIMF * DIMF, xn);
        float* tx = xc; xc = xn; xn = tx;
    }
    k_zero<<<blks(N_RES * DIMF, 256), 256, 0, stream>>>(fres, N_RES * DIMF);
    k_res_scatter<<<blks(NA * DIMF, 256), 256, 0, stream>>>(xc, r2a, fres);

    // ---- projections ----
    k_proj1<<<blks(N_RES * DIMF, 256), 256, 0, stream>>>(fcur, p1w, p1b, Fj);
    k_proj2<<<blks(N_RES * DIMF, 256), 256, 0, stream>>>(Fj, p2w, p2b, mp);
    k_proj1<<<blks(N_RES * DIMF, 256), 256, 0, stream>>>(fres, p1w, p1b, Fj);
    k_proj2<<<blks(N_RES * DIMF, 256), 256, 0, stream>>>(Fj, p2w, p2b, sp);

    // ---- mask, sim, losses ----
    k_zero<<<blks(N_RES * N_RES, 256), 256, 0, stream>>>(maskb, N_RES * N_RES);
    k_mask<<<N_RES, 32, 0, stream>>>(adj, maskb);
    k_norms<<<blks(2 * N_RES, 256), 256, 0, stream>>>(mp, sp, norms);
    k_sim<<<blks(N_RES * N_RES, 256), 256, 0, stream>>>(mp, sp, norms, simb);
    k_zero<<<1, 32, 0, stream>>>(out + N_RES, 1);
    k_loss<<<N_RES, 128, 0, stream>>>(simb, maskb, out + N_RES);

    // ---- attention head + MLP -> y ----
    k_head<<<N_RES, 256, 0, stream>>>(mp, sp, kw, gw, gb,
                                      m1w, m1b, m2w, m2b, m3w, m3b, out);
}
